// FrenchToEnglish_59253368816066
// MI455X (gfx1250) — compile-verified, matrix operand order checked by reading, not run on hardware
//
#include <hip/hip_runtime.h>
#include <hip/hip_bf16.h>
#include <stdint.h>

// ---------------- model constants ----------------
#define MB       32          // batch
#define MS       128         // source length
#define ME       300         // embed dim
#define MEP      320         // embed dim padded to mult of 32
#define MH       1024        // hidden
#define M3H      3072        // 3*H
#define MV       32000       // vocab
#define MLEN     48          // decode length
#define EN_BOS_T 1

// ---------------- WMMA types / helpers ----------------
typedef __bf16 v16bf __attribute__((ext_vector_type(16)));
typedef float  v8f   __attribute__((ext_vector_type(8)));

static __device__ __forceinline__ unsigned short f2bf(float x) {
    unsigned u = __builtin_bit_cast(unsigned, x);
    u = (u + 0x7FFFu + ((u >> 16) & 1u)) >> 16;   // round-to-nearest-even
    return (unsigned short)u;
}

// Load one 16x32 bf16 A/B fragment for v_wmma_f32_16x16x32_bf16.
// Per ISA layout: lanes 0-15 hold K={0..7,16..23}, lanes 16-31 hold K={8..15,24..31},
// row/col = lane&15.  bf16 pairs are contiguous -> two 16B vector loads.
static __device__ __forceinline__ v16bf load_frag(const unsigned short* __restrict__ base,
                                                  int ld, int row, int k0, int half) {
    union { v16bf v; uint4 q[2]; } f;
    f.q[0] = *(const uint4*)(base + (size_t)row * ld + k0 + 8 * half);
    f.q[1] = *(const uint4*)(base + (size_t)row * ld + k0 + 16 + 8 * half);
    return f.v;
}

static __device__ __forceinline__ v8f wmma_bf16(v16bf a, v16bf b, v8f c) {
    return __builtin_amdgcn_wmma_f32_16x16x32_bf16(false, a, false, b, (short)0, c, false, false);
}

static __device__ __forceinline__ float sigmoidf_(float x) {
    return 1.0f / (1.0f + expf(-x));
}

// ---------------- kernels ----------------

// fp32 [N,K] -> bf16 [N,Kp], zero-padded in K.
__global__ void cvt_pad_bf16(const float* __restrict__ src, unsigned short* __restrict__ dst,
                             int N, int K, int Kp) {
    size_t idx = (size_t)blockIdx.x * blockDim.x + threadIdx.x;
    size_t total = (size_t)N * Kp;
    if (idx >= total) return;
    int col = (int)(idx % Kp);
    size_t row = idx / Kp;
    dst[idx] = (col < K) ? f2bf(src[row * (size_t)K + col]) : (unsigned short)0;
}

// Gather encoder inputs: X[s*B+b, e] = bf16(fr_emb[fr[b,s], e]), padded to MEP.
__global__ void gather_fr_kernel(const long long* __restrict__ fr,
                                 const float* __restrict__ fr_emb,
                                 unsigned short* __restrict__ X) {
    size_t idx = (size_t)blockIdx.x * blockDim.x + threadIdx.x;
    size_t total = (size_t)MS * MB * MEP;
    if (idx >= total) return;
    int e = (int)(idx % MEP);
    size_t row = idx / MEP;
    int s = (int)(row / MB), b = (int)(row % MB);
    long long tok = fr[(size_t)b * MS + s];
    X[idx] = (e < ME) ? f2bf(fr_emb[(size_t)tok * ME + e]) : (unsigned short)0;
}

// C[m, n] = sum_k A[m,k]*W[n,k] + bias[n].
// A: [M,K] bf16 packed; W: [N,K] bf16 packed; C fp32 with row stride ldc.
// Block = 128 threads (4 waves).  The 32xK A-panel is staged in LDS once per block;
// each wave owns NT=4 n-tiles (block covers 256 columns) and reuses the LDS A-frags
// across all of them: per K-chunk -> 2 LDS A loads + 4 global B loads + 8 WMMAs.
// grid.x = N/256, grid.y = M/32.
template <int NT>
__global__ void gemm_bf16_wmma(const unsigned short* __restrict__ A,
                               const unsigned short* __restrict__ W, int ldw,
                               const float* __restrict__ bias,
                               float* __restrict__ C, long long ldc, int K) {
    __shared__ unsigned short sA[32 * MH];   // up to 64 KB (K <= 1024)

    int tid  = threadIdx.x;
    int lane = tid & 31;
    int wave = tid >> 5;
    int m0   = blockIdx.y * 32;
    int n0   = (blockIdx.x * 4 + wave) * (16 * NT);
    int half = lane >> 4;
    int l15  = lane & 15;

    // stage the 32-row A panel (packed, so it is one linear block)
    {
        const uint4* src = (const uint4*)(A + (size_t)m0 * K);
        uint4* dst = (uint4*)sA;
        int total = 32 * K / 8;
        for (int i = tid; i < total; i += 128) dst[i] = src[i];
    }
    __syncthreads();

    v8f acc0[NT], acc1[NT];
#pragma unroll
    for (int j = 0; j < NT; ++j) { acc0[j] = (v8f){}; acc1[j] = (v8f){}; }

    for (int k0 = 0; k0 < K; k0 += 32) {
        if (k0 + 32 < K)  // warm L2 for the next weight chunk
            __builtin_prefetch((const void*)(W + (size_t)(n0 + l15) * ldw + k0 + 32), 0, 1);
        v16bf a0 = load_frag(sA, K, l15,      k0, half);
        v16bf a1 = load_frag(sA, K, 16 + l15, k0, half);
#pragma unroll
        for (int j = 0; j < NT; ++j) {
            v16bf b = load_frag(W, ldw, n0 + j * 16 + l15, k0, half);
            acc0[j] = wmma_bf16(a0, b, acc0[j]);
            acc1[j] = wmma_bf16(a1, b, acc1[j]);
        }
    }

#pragma unroll
    for (int j = 0; j < NT; ++j) {
        int n = n0 + j * 16 + l15;
        float bv = bias ? bias[n] : 0.0f;
#pragma unroll
        for (int v = 0; v < 8; ++v) {
            int m = v + 8 * half;
            C[(size_t)(m0 + m)      * ldc + n] = acc0[j][v] + bv;
            C[(size_t)(m0 + 16 + m) * ldc + n] = acc1[j][v] + bv;
        }
    }
}

// Fused GRU recurrence step (one timestep, whole batch of 32, one hidden-tile of 16
// units per wave).  h (bf16) is staged in LDS per block; gh = h@Whh^T for the r/z/n
// gate rows is computed with WMMA, then all gate math happens in registers (the
// 16x16 C layout keeps r/z/n for the same (m, unit) in the same lane/element).
// gi = x@Wih^T + bih is precomputed.  grid = 16 blocks x 128 threads.
__global__ void gru_step_wmma(const unsigned short* __restrict__ h_bf,  // [32,H]
                              const float* __restrict__ h_f,            // [32,H]
                              const unsigned short* __restrict__ Whh,   // [3H,H] bf16
                              const float* __restrict__ bhh,            // [3H]
                              const float* __restrict__ gi,             // [32,3H]
                              float* __restrict__ ho_f,                 // [32,H]
                              unsigned short* __restrict__ ho_bf,       // [32,H]
                              unsigned short* __restrict__ ys_bf) {     // optional [32,H]
    __shared__ unsigned short sH[32 * MH];   // 64 KB

    int tid  = threadIdx.x;
    int lane = tid & 31;
    int wave = tid >> 5;
    int tile = blockIdx.x * 4 + wave;   // 0..63
    int i0   = tile * 16;
    int half = lane >> 4;
    int l15  = lane & 15;

    {
        const uint4* src = (const uint4*)h_bf;
        uint4* dst = (uint4*)sH;
        int total = 32 * MH / 8;
        for (int i = tid; i < total; i += 128) dst[i] = src[i];
    }
    __syncthreads();

    v8f ar0 = {}, ar1 = {}, az0 = {}, az1 = {}, an0 = {}, an1 = {};
    for (int k0 = 0; k0 < MH; k0 += 32) {
        v16bf a0 = load_frag(sH, MH, l15,      k0, half);
        v16bf a1 = load_frag(sH, MH, 16 + l15, k0, half);
        v16bf br = load_frag(Whh, MH, 0 * MH + i0 + l15, k0, half);
        v16bf bz = load_frag(Whh, MH, 1 * MH + i0 + l15, k0, half);
        v16bf bn = load_frag(Whh, MH, 2 * MH + i0 + l15, k0, half);
        ar0 = wmma_bf16(a0, br, ar0);  ar1 = wmma_bf16(a1, br, ar1);
        az0 = wmma_bf16(a0, bz, az0);  az1 = wmma_bf16(a1, bz, az1);
        an0 = wmma_bf16(a0, bn, an0);  an1 = wmma_bf16(a1, bn, an1);
    }

    int col = i0 + l15;
    float br_b = bhh[col], bz_b = bhh[MH + col], bn_b = bhh[2 * MH + col];
#pragma unroll
    for (int mh = 0; mh < 2; ++mh) {
#pragma unroll
        for (int v = 0; v < 8; ++v) {
            int m = mh * 16 + v + 8 * half;
            float hr = (mh ? ar1[v] : ar0[v]) + br_b;
            float hz = (mh ? az1[v] : az0[v]) + bz_b;
            float hn = (mh ? an1[v] : an0[v]) + bn_b;
            size_t g = (size_t)m * M3H;
            float r  = sigmoidf_(gi[g + col] + hr);
            float z  = sigmoidf_(gi[g + MH + col] + hz);
            float nn = tanhf(gi[g + 2 * MH + col] + r * hn);
            float hp = h_f[(size_t)m * MH + col];
            float hv = (1.0f - z) * nn + z * hp;
            ho_f[(size_t)m * MH + col] = hv;
            unsigned short hb = f2bf(hv);
            ho_bf[(size_t)m * MH + col] = hb;
            if (ys_bf) ys_bf[(size_t)m * MH + col] = hb;
        }
    }
}

// Per-batch-row argmax over 32000 logits (first-max tie-break like jnp.argmax),
// then gather en_emb[tok] into the padded bf16 embedding buffer for the next step.
// fixed_tok >= 0 -> skip argmax (BOS init).
__global__ void argmax_embed(const float* __restrict__ logits, long long row_stride,
                             const float* __restrict__ en_emb,
                             unsigned short* __restrict__ emb_out,  // [32,MEP]
                             int fixed_tok) {
    __shared__ float smax[256];
    __shared__ int   sidx[256];
    int b = blockIdx.x, tid = threadIdx.x;
    int best = 0;
    if (fixed_tok < 0) {
        const float* row = logits + (size_t)b * row_stride;
        float bv = -INFINITY; int bi = 0;
        for (int i = tid; i < MV; i += 256) {
            float x = row[i];
            if (x > bv) { bv = x; bi = i; }
        }
        smax[tid] = bv; sidx[tid] = bi;
        __syncthreads();
        for (int s = 128; s > 0; s >>= 1) {
            if (tid < s) {
                float xo = smax[tid + s]; int io = sidx[tid + s];
                if (xo > smax[tid] || (xo == smax[tid] && io < sidx[tid])) {
                    smax[tid] = xo; sidx[tid] = io;
                }
            }
            __syncthreads();
        }
        best = sidx[0];
    } else {
        best = fixed_tok;
    }
    __syncthreads();
    for (int e = tid; e < MEP; e += 256) {
        float val = (e < ME) ? en_emb[(size_t)best * ME + e] : 0.0f;
        emb_out[(size_t)b * MEP + e] = f2bf(val);
    }
}

// ---------------- host orchestration ----------------
extern "C" void kernel_launch(void* const* d_in, const int* in_sizes, int n_in,
                              void* d_out, int out_size, void* d_ws, size_t ws_size,
                              hipStream_t stream) {
    (void)in_sizes; (void)n_in; (void)out_size; (void)ws_size;

    const long long* fr     = (const long long*)d_in[0];
    const float* fr_emb     = (const float*)d_in[1];
    const float* en_emb     = (const float*)d_in[2];
    const float* enc_Wih0   = (const float*)d_in[3];
    const float* enc_Whh0   = (const float*)d_in[4];
    const float* enc_bih0   = (const float*)d_in[5];
    const float* enc_bhh0   = (const float*)d_in[6];
    const float* enc_Wih1   = (const float*)d_in[7];
    const float* enc_Whh1   = (const float*)d_in[8];
    const float* enc_bih1   = (const float*)d_in[9];
    const float* enc_bhh1   = (const float*)d_in[10];
    const float* trans_W    = (const float*)d_in[11];
    const float* trans_b    = (const float*)d_in[12];
    const float* dec_Wih0   = (const float*)d_in[13];
    const float* dec_Whh0   = (const float*)d_in[14];
    const float* dec_bih0   = (const float*)d_in[15];
    const float* dec_bhh0   = (const float*)d_in[16];
    const float* dec_Wih1   = (const float*)d_in[17];
    const float* dec_Whh1   = (const float*)d_in[18];
    const float* dec_bih1   = (const float*)d_in[19];
    const float* dec_bhh1   = (const float*)d_in[20];
    const float* vocab_W    = (const float*)d_in[21];
    const float* vocab_b    = (const float*)d_in[22];
    float* out              = (float*)d_out;

    char* ws = (char*)d_ws;
    size_t off = 0;
    auto alloc = [&](size_t bytes) -> void* {
        void* p = ws + off;
        off = (off + bytes + 255) & ~(size_t)255;
        return p;
    };

    // bf16 weight mirrors (L2-resident working set)
    unsigned short* eWih0 = (unsigned short*)alloc((size_t)M3H * MEP * 2);
    unsigned short* eWhh0 = (unsigned short*)alloc((size_t)M3H * MH * 2);
    unsigned short* eWih1 = (unsigned short*)alloc((size_t)M3H * MH * 2);
    unsigned short* eWhh1 = (unsigned short*)alloc((size_t)M3H * MH * 2);
    unsigned short* dWih0 = (unsigned short*)alloc((size_t)M3H * MEP * 2);
    unsigned short* dWhh0 = (unsigned short*)alloc((size_t)M3H * MH * 2);
    unsigned short* dWih1 = (unsigned short*)alloc((size_t)M3H * MH * 2);
    unsigned short* dWhh1 = (unsigned short*)alloc((size_t)M3H * MH * 2);
    unsigned short* tWb   = (unsigned short*)alloc((size_t)MH * MH * 2);
    unsigned short* vWb   = (unsigned short*)alloc((size_t)MV * MH * 2);
    // activations
    unsigned short* Xenc  = (unsigned short*)alloc((size_t)MS * MB * MEP * 2);
    float* gi0_all        = (float*)alloc((size_t)MS * MB * M3H * 4);
    float* gi1_all        = (float*)alloc((size_t)MS * MB * M3H * 4);
    unsigned short* ys0   = (unsigned short*)alloc((size_t)MS * MB * MH * 2);
    float* hf[8]; unsigned short* hb[8];
    for (int i = 0; i < 8; ++i) hf[i] = (float*)alloc((size_t)MB * MH * 4);
    for (int i = 0; i < 8; ++i) hb[i] = (unsigned short*)alloc((size_t)MB * MH * 2);
    float* gid0           = (float*)alloc((size_t)MB * M3H * 4);
    float* gid1           = (float*)alloc((size_t)MB * M3H * 4);
    unsigned short* embd  = (unsigned short*)alloc((size_t)MB * MEP * 2);

    auto cvt = [&](const float* s, unsigned short* d, int N, int K, int Kp) {
        size_t total = (size_t)N * Kp;
        cvt_pad_bf16<<<dim3((unsigned)((total + 255) / 256)), dim3(256), 0, stream>>>(s, d, N, K, Kp);
    };

    // 1) weight conversion
    cvt(enc_Wih0, eWih0, M3H, ME, MEP);
    cvt(enc_Whh0, eWhh0, M3H, MH, MH);
    cvt(enc_Wih1, eWih1, M3H, MH, MH);
    cvt(enc_Whh1, eWhh1, M3H, MH, MH);
    cvt(dec_Wih0, dWih0, M3H, ME, MEP);
    cvt(dec_Whh0, dWhh0, M3H, MH, MH);
    cvt(dec_Wih1, dWih1, M3H, MH, MH);
    cvt(dec_Whh1, dWhh1, M3H, MH, MH);
    cvt(trans_W,  tWb,   MH,  MH, MH);
    cvt(vocab_W,  vWb,   MV,  MH, MH);

    // 2) gather encoder inputs (time-major, bf16, padded)
    {
        size_t total = (size_t)MS * MB * MEP;
        gather_fr_kernel<<<dim3((unsigned)((total + 255) / 256)), dim3(256), 0, stream>>>(fr, fr_emb, Xenc);
    }

    // 3) encoder layer-0 input projection for ALL timesteps in one GEMM
    gemm_bf16_wmma<4><<<dim3(M3H / 256, MS * MB / 32), dim3(128), 0, stream>>>(
        Xenc, eWih0, MEP, enc_bih0, gi0_all, (long long)M3H, MEP);

    // 4) encoder layer-0 recurrence
    hipMemsetAsync(hf[0], 0, (size_t)MB * MH * 4, stream);
    hipMemsetAsync(hb[0], 0, (size_t)MB * MH * 2, stream);
    float* hcf = hf[0]; float* hnf = hf[1];
    unsigned short* hcb = hb[0]; unsigned short* hnb = hb[1];
    for (int s = 0; s < MS; ++s) {
        gru_step_wmma<<<dim3(16), dim3(128), 0, stream>>>(
            hcb, hcf, eWhh0, enc_bhh0, gi0_all + (size_t)s * MB * M3H,
            hnf, hnb, ys0 + (size_t)s * MB * MH);
        float* tf = hcf; hcf = hnf; hnf = tf;
        unsigned short* tb = hcb; hcb = hnb; hnb = tb;
    }
    unsigned short* hT0_bf = hcb;   // final layer-0 hidden (bf16)

    // 5) encoder layer-1 input projection (all timesteps) + recurrence
    gemm_bf16_wmma<4><<<dim3(M3H / 256, MS * MB / 32), dim3(128), 0, stream>>>(
        ys0, eWih1, MH, enc_bih1, gi1_all, (long long)M3H, MH);
    hipMemsetAsync(hf[2], 0, (size_t)MB * MH * 4, stream);
    hipMemsetAsync(hb[2], 0, (size_t)MB * MH * 2, stream);
    hcf = hf[2]; hnf = hf[3]; hcb = hb[2]; hnb = hb[3];
    for (int s = 0; s < MS; ++s) {
        gru_step_wmma<<<dim3(16), dim3(128), 0, stream>>>(
            hcb, hcf, eWhh1, enc_bhh1, gi1_all + (size_t)s * MB * M3H,
            hnf, hnb, (unsigned short*)nullptr);
        float* tf = hcf; hcf = hnf; hnf = tf;
        unsigned short* tb = hcb; hcb = hnb; hnb = tb;
    }
    unsigned short* hT1_bf = hcb;   // final layer-1 hidden (bf16)

    // 6) hidden transform: hid = hT @ trans_W^T + trans_b  (decoder init states)
    float* d0f_c = hf[4]; float* d0f_n = hf[5];
    float* d1f_c = hf[6]; float* d1f_n = hf[7];
    unsigned short* d0b_c = hb[4]; unsigned short* d0b_n = hb[5];
    unsigned short* d1b_c = hb[6]; unsigned short* d1b_n = hb[7];
    gemm_bf16_wmma<4><<<dim3(MH / 256, 1), dim3(128), 0, stream>>>(
        hT0_bf, tWb, MH, trans_b, d0f_c, (long long)MH, MH);
    gemm_bf16_wmma<4><<<dim3(MH / 256, 1), dim3(128), 0, stream>>>(
        hT1_bf, tWb, MH, trans_b, d1f_c, (long long)MH, MH);
    cvt(d0f_c, d0b_c, MB, MH, MH);
    cvt(d1f_c, d1b_c, MB, MH, MH);

    // 7) BOS embedding
    argmax_embed<<<dim3(MB), dim3(256), 0, stream>>>(
        (const float*)nullptr, 0, en_emb, embd, EN_BOS_T);

    // 8) autoregressive decode, MAX_LEN greedy steps
    const long long ldo = (long long)MLEN * MV;   // d_out row stride per batch
    for (int t = 0; t < MLEN; ++t) {
        gemm_bf16_wmma<4><<<dim3(M3H / 256, 1), dim3(128), 0, stream>>>(
            embd, dWih0, MEP, dec_bih0, gid0, (long long)M3H, MEP);
        gru_step_wmma<<<dim3(16), dim3(128), 0, stream>>>(
            d0b_c, d0f_c, dWhh0, dec_bhh0, gid0, d0f_n, d0b_n, (unsigned short*)nullptr);
        gemm_bf16_wmma<4><<<dim3(M3H / 256, 1), dim3(128), 0, stream>>>(
            d0b_n, dWih1, MH, dec_bih1, gid1, (long long)M3H, MH);
        gru_step_wmma<<<dim3(16), dim3(128), 0, stream>>>(
            d1b_c, d1f_c, dWhh1, dec_bhh1, gid1, d1f_n, d1b_n, (unsigned short*)nullptr);
        // logits straight into d_out[b, t, :]
        gemm_bf16_wmma<4><<<dim3(MV / 256, 1), dim3(128), 0, stream>>>(
            d1b_n, vWb, MH, vocab_b, out + (size_t)t * MV, ldo, MH);
        // greedy token -> next embedding
        argmax_embed<<<dim3(MB), dim3(256), 0, stream>>>(
            out + (size_t)t * MV, ldo, en_emb, embd, -1);
        // swap decoder hidden ping-pongs
        float* tf = d0f_c; d0f_c = d0f_n; d0f_n = tf;
        unsigned short* tb = d0b_c; d0b_c = d0b_n; d0b_n = tb;
        tf = d1f_c; d1f_c = d1f_n; d1f_n = tf;
        tb = d1b_c; d1b_c = d1b_n; d1b_n = tb;
    }
}
